// Attention_3891240370491
// MI455X (gfx1250) — matrix operand order repaired and checked by reading, not compiled, not run on hardware
//
#include <hip/hip_runtime.h>

// ---------------------------------------------------------------------------
// Problem constants (from reference)
// ---------------------------------------------------------------------------
#define BATCH   4
#define SEQ     2048
#define DMODEL  1024
#define HEADS   16
#define DHEAD   64
#define INNER   1024          // HEADS*DHEAD
#define QKV_N   3072          // 3*INNER
#define NROWS   (BATCH*SEQ)   // 8192

typedef __bf16 bf16;
typedef bf16  v16bf __attribute__((ext_vector_type(16)));
typedef float v8f   __attribute__((ext_vector_type(8)));

// ---------------------------------------------------------------------------
// WMMA helpers (V_WMMA_F32_16X16X32_BF16, D = A(16x32) * B(32x16) + C)
// ---------------------------------------------------------------------------
__device__ __forceinline__ v8f wmma_bf16(v16bf a, v16bf b, v8f c) {
  return __builtin_amdgcn_wmma_f32_16x16x32_bf16(
      /*neg_a=*/false, a, /*neg_b=*/false, b,
      /*c_mod=*/(short)0, c, /*reuse_a=*/false, /*reuse_b=*/false);
}

// A-matrix 16x32 bf16 fragment. Lane L holds row m = L%16; element j maps to
// K = (j>>3)*16 + (L>>4)*8 + (j&7)   (ISA 7.12.2 "16-bit A-Matrix 16x32").
__device__ __forceinline__ v16bf load_fragA(const bf16* __restrict__ p, int lane) {
  const int hs = (lane >> 4) << 3;   // 0 or 8
  union { v16bf v; uint4 u[2]; } r;
  r.u[0] = *reinterpret_cast<const uint4*>(p + hs);
  r.u[1] = *reinterpret_cast<const uint4*>(p + 16 + hs);
  return r.v;
}

// B-matrix 32x16 bf16 fragment. Lane L holds column n = L%16; element j maps
// to K = (j>>3)*16 + (L>>4)*8 + (j&7).  Source "column" must be contiguous over K.
__device__ __forceinline__ v16bf load_fragB(const bf16* __restrict__ p, int lane) {
  union { v16bf v; uint4 u[2]; } r;
  const int hs = (lane >> 4) << 3;
  r.u[0] = *reinterpret_cast<const uint4*>(p + hs);
  r.u[1] = *reinterpret_cast<const uint4*>(p + 16 + hs);
  return r.v;
}

// ---------------------------------------------------------------------------
// CDNA5 async copy: GLOBAL_LOAD_ASYNC_TO_LDS_B128 (ASYNCcnt-tracked DMA).
// lds_off = low 32 bits of a generic LDS pointer (== LDS byte offset per the
// aperture rule, ISA 10.2).  Wait via s_wait_asynccnt.
// ---------------------------------------------------------------------------
__device__ __forceinline__ void async_copy_b128(void* lds_ptr, const void* gptr) {
  unsigned lds_off = (unsigned)(unsigned long long)lds_ptr;
  unsigned long long ga = (unsigned long long)gptr;
  asm volatile("global_load_async_to_lds_b128 %0, %1, off"
               :: "v"(lds_off), "v"(ga) : "memory");
}
__device__ __forceinline__ void wait_async0() {
  asm volatile("s_wait_asynccnt 0x0" ::: "memory");
}

// ---------------------------------------------------------------------------
// Kernel 1: fp32 -> bf16 convert (grid-stride)
// ---------------------------------------------------------------------------
__global__ void k_cvt_bf16(const float* __restrict__ src, bf16* __restrict__ dst, int n) {
  int i = blockIdx.x * blockDim.x + threadIdx.x;
  int stride = gridDim.x * blockDim.x;
  for (; i < n; i += stride) dst[i] = (bf16)src[i];
}

// ---------------------------------------------------------------------------
// Kernel 2: QKV GEMM  qkv[row,col] = x[row,:] . w_qkv[col,:] + b[col]
// Block = 128 rows x 64 cols (8 waves x 16 rows).  The 64x32 bf16 weight tile
// is staged in LDS once per block via async DMA (double-buffered), shared by
// all 8 waves; A fragments stream from global (L2-resident).
// ---------------------------------------------------------------------------
__global__ __launch_bounds__(256) void k_qkv_gemm(
    const bf16* __restrict__ xb, const bf16* __restrict__ wb,
    const float* __restrict__ bias,
    float* __restrict__ qf, float* __restrict__ kf, float* __restrict__ vf) {
  __shared__ __align__(16) bf16 bsm[2][64 * 32];   // [buf][col*32 + k], 4 KB each
  const int tid  = threadIdx.x;
  const int lane = tid & 31;
  const int wave = tid >> 5;
  const int l16  = lane & 15;
  const int cgroups = QKV_N / 64;                  // 48
  const int rowblk  = blockIdx.x / cgroups;        // 0..63 (128 rows each)
  const int colbase = (blockIdx.x % cgroups) * 64;

  // Async-copy mapping: thread t moves 16 B; col = t>>2, k-segment = (t&3)*8.
  const int ccol = tid >> 2;
  const int kseg = (tid & 3) * 8;
  const bf16* wsrc = wb + (size_t)(colbase + ccol) * DMODEL + kseg;
  bf16* ldst0 = &bsm[0][ccol * 32 + kseg];
  bf16* ldst1 = &bsm[1][ccol * 32 + kseg];

  async_copy_b128(ldst0, wsrc);                    // preload chunk 0

  const bf16* arow = xb + (size_t)(rowblk * 128 + wave * 16 + l16) * DMODEL;
  v8f acc[4] = {};
  int p = 0;
  for (int kk = 0; kk < DMODEL; kk += 32) {
    wait_async0();        // own async copy into bsm[p] has landed
    __syncthreads();      // everyone's landed; everyone's reads of bsm[p^1] done
    if (kk + 32 < DMODEL)
      async_copy_b128(p ? ldst0 : ldst1, wsrc + kk + 32);

    v16bf a = load_fragA(arow + kk, lane);
    const bf16* bbase = &bsm[p][0];
#pragma unroll
    for (int f = 0; f < 4; ++f) {
      v16bf b = load_fragB(bbase + (f * 16 + l16) * 32, lane);  // ds_load_b128
      acc[f] = wmma_bf16(a, b, acc[f]);
    }
    p ^= 1;
  }

  // Epilogue: C/D layout rows are i + 8*(lane>>4); col = lane%16 (constant).
  const int rbase = rowblk * 128 + wave * 16 + ((lane >> 4) << 3);
  const int b_ = rbase >> 11;
  const int nbase = rbase & 2047;
#pragma unroll
  for (int f = 0; f < 4; ++f) {
    const int col = colbase + f * 16 + l16;
    const float bv = bias[col];
    const int s = col >> 10, h = (col >> 6) & 15, d = col & 63;
    float* dst = (s == 0) ? qf : (s == 1) ? kf : vf;
    const size_t base = ((size_t)(b_ * HEADS + h) * SEQ + nbase) * DHEAD + d;
#pragma unroll
    for (int i = 0; i < 8; ++i)
      dst[base + (size_t)i * DHEAD] = acc[f][i] + bv;
  }
}

// ---------------------------------------------------------------------------
// Kernel 3: per-row L2 normalize q,k (head scale folded into q), pack bf16,
//           and emit v transposed [B,H,64,N].  One wave per (b,h,n) row.
// ---------------------------------------------------------------------------
__global__ __launch_bounds__(256) void k_norm_pack(
    const float* __restrict__ qf, const float* __restrict__ kf,
    const float* __restrict__ vf, const float* __restrict__ logit_scale,
    bf16* __restrict__ qb, bf16* __restrict__ kb, bf16* __restrict__ vT) {
  const int lane = threadIdx.x & 31;
  const long row = (long)blockIdx.x * 8 + (threadIdx.x >> 5); // [bh][n] flat
  if (row >= (long)BATCH * HEADS * SEQ) return;
  const int bh = (int)(row >> 11);
  const int h  = bh & (HEADS - 1);
  const int n  = (int)(row & (SEQ - 1));
  const size_t off = (size_t)row * DHEAD;

  const float sc = __expf(fminf(logit_scale[h], 4.60517018598809136804f)); // ln(100)

  float q0 = qf[off + lane * 2], q1 = qf[off + lane * 2 + 1];
  float ss = q0 * q0 + q1 * q1;
#pragma unroll
  for (int m = 16; m >= 1; m >>= 1) ss += __shfl_xor(ss, m, 32);
  float inv = sc / fmaxf(__builtin_sqrtf(ss), 1e-12f);
  qb[off + lane * 2]     = (bf16)(q0 * inv);
  qb[off + lane * 2 + 1] = (bf16)(q1 * inv);

  float k0 = kf[off + lane * 2], k1 = kf[off + lane * 2 + 1];
  float sk = k0 * k0 + k1 * k1;
#pragma unroll
  for (int m = 16; m >= 1; m >>= 1) sk += __shfl_xor(sk, m, 32);
  float invk = 1.0f / fmaxf(__builtin_sqrtf(sk), 1e-12f);
  kb[off + lane * 2]     = (bf16)(k0 * invk);
  kb[off + lane * 2 + 1] = (bf16)(k1 * invk);

  const size_t tb = (size_t)bh * DHEAD * SEQ;
  vT[tb + (size_t)(lane * 2) * SEQ + n]     = (bf16)vf[off + lane * 2];
  vT[tb + (size_t)(lane * 2 + 1) * SEQ + n] = (bf16)vf[off + lane * 2 + 1];
}

// ---------------------------------------------------------------------------
// Kernel 4: flash attention, one wave per (b,h, 16-row q tile).
// ---------------------------------------------------------------------------
__global__ __launch_bounds__(32) void k_attn(
    const bf16* __restrict__ qb, const bf16* __restrict__ kb,
    const bf16* __restrict__ vT, bf16* __restrict__ ctx) {
  __shared__ __align__(16) bf16 lds[16][32];
  const int lane = threadIdx.x;
  const int l16  = lane & 15;
  const int bh    = blockIdx.x >> 7;              // SEQ/16 = 128 q-tiles per bh
  const int qtile = blockIdx.x & 127;
  const int b_ = bh >> 4, h = bh & 15;

  const bf16* qrow = qb + ((size_t)bh * SEQ + qtile * 16 + l16) * DHEAD;
  const v16bf qa0 = load_fragA(qrow, lane);        // d = 0..31
  const v16bf qa1 = load_fragA(qrow + 32, lane);   // d = 32..63

  float mrun[8], lrun[8];
  v8f acc[4] = {};
#pragma unroll
  for (int i = 0; i < 8; ++i) { mrun[i] = -1e30f; lrun[i] = 0.0f; }

  const bf16* kbbase = kb + (size_t)bh * SEQ * DHEAD;
  const bf16* vtbase = vT + (size_t)bh * DHEAD * SEQ;

  for (int key = 0; key < SEQ; key += 32) {
    v8f s[2];
#pragma unroll
    for (int t = 0; t < 2; ++t) {
      const bf16* krow = kbbase + (size_t)(key + t * 16 + l16) * DHEAD;
      v16bf b0 = load_fragB(krow, lane);
      v16bf b1 = load_fragB(krow + 32, lane);
      v8f c = {};
      c = wmma_bf16(qa0, b0, c);
      s[t] = wmma_bf16(qa1, b1, c);
    }
    if (key + 32 < SEQ)   // prefetch next K chunk (global_prefetch_b8)
      __builtin_prefetch(kbbase + (size_t)(key + 32 + l16) * DHEAD, 0, 1);

    float mnew[8], corr[8];
#pragma unroll
    for (int i = 0; i < 8; ++i) {
      float mx = fmaxf(s[0][i], s[1][i]);
      mx = fmaxf(mx, __shfl_xor(mx, 1, 16));
      mx = fmaxf(mx, __shfl_xor(mx, 2, 16));
      mx = fmaxf(mx, __shfl_xor(mx, 4, 16));
      mx = fmaxf(mx, __shfl_xor(mx, 8, 16));
      mnew[i] = fmaxf(mrun[i], mx);
      corr[i] = __expf(mrun[i] - mnew[i]);
      mrun[i] = mnew[i];
    }
    const int mrow = (lane >> 4) << 3;
#pragma unroll
    for (int t = 0; t < 2; ++t)
#pragma unroll
      for (int i = 0; i < 8; ++i) {
        float p = __expf(s[t][i] - mnew[i]);
        s[t][i] = p;
        lds[mrow + i][t * 16 + l16] = (bf16)p;   // 16x32 P tile, row-major
      }
#pragma unroll
    for (int i = 0; i < 8; ++i) {
      float rs = s[0][i] + s[1][i];
      rs += __shfl_xor(rs, 1, 16);
      rs += __shfl_xor(rs, 2, 16);
      rs += __shfl_xor(rs, 4, 16);
      rs += __shfl_xor(rs, 8, 16);
      lrun[i] = lrun[i] * corr[i] + rs;
    }
#pragma unroll
    for (int f = 0; f < 4; ++f)
#pragma unroll
      for (int i = 0; i < 8; ++i) acc[f][i] *= corr[i];

    __syncthreads();   // single-wave WG: barrier is S_NOP, keeps the dscnt fence

    v16bf pa = load_fragA(&lds[l16][0], lane);
#pragma unroll
    for (int f = 0; f < 4; ++f) {
      const bf16* vrow = vtbase + (size_t)(f * 16 + l16) * SEQ + key;
      v16bf vb = load_fragB(vrow, lane);
      acc[f] = wmma_bf16(pa, vb, acc[f]);
    }
  }

  const int nbase = qtile * 16 + ((lane >> 4) << 3);
#pragma unroll
  for (int f = 0; f < 4; ++f) {
#pragma unroll
    for (int i = 0; i < 8; ++i) {
      float o = acc[f][i] / fmaxf(lrun[i], 1e-30f);
      ctx[((size_t)(b_ * SEQ + nbase + i)) * INNER + h * DHEAD + f * 16 + l16] = (bf16)o;
    }
  }
}

// ---------------------------------------------------------------------------
// Kernel 5: output GEMM  y[row,col] = ctx[row,:] . w_out[col,:]   (fp32 out)
// Same async-LDS double-buffered staging of the weight tile as k_qkv_gemm.
// ---------------------------------------------------------------------------
__global__ __launch_bounds__(256) void k_out_gemm(
    const bf16* __restrict__ cb, const bf16* __restrict__ wb,
    float* __restrict__ out) {
  __shared__ __align__(16) bf16 bsm[2][64 * 32];
  const int tid  = threadIdx.x;
  const int lane = tid & 31;
  const int wave = tid >> 5;
  const int l16  = lane & 15;
  const int cgroups = DMODEL / 64;                 // 16
  const int rowblk  = blockIdx.x / cgroups;        // 0..63
  const int colbase = (blockIdx.x % cgroups) * 64;

  const int ccol = tid >> 2;
  const int kseg = (tid & 3) * 8;
  const bf16* wsrc = wb + (size_t)(colbase + ccol) * DMODEL + kseg;
  bf16* ldst0 = &bsm[0][ccol * 32 + kseg];
  bf16* ldst1 = &bsm[1][ccol * 32 + kseg];

  async_copy_b128(ldst0, wsrc);

  const bf16* arow = cb + (size_t)(rowblk * 128 + wave * 16 + l16) * DMODEL;
  v8f acc[4] = {};
  int p = 0;
  for (int kk = 0; kk < DMODEL; kk += 32) {
    wait_async0();
    __syncthreads();
    if (kk + 32 < DMODEL)
      async_copy_b128(p ? ldst0 : ldst1, wsrc + kk + 32);

    v16bf a = load_fragA(arow + kk, lane);
    const bf16* bbase = &bsm[p][0];
#pragma unroll
    for (int f = 0; f < 4; ++f) {
      v16bf b = load_fragB(bbase + (f * 16 + l16) * 32, lane);
      acc[f] = wmma_bf16(a, b, acc[f]);
    }
    p ^= 1;
  }

  const int rbase = rowblk * 128 + wave * 16 + ((lane >> 4) << 3);
#pragma unroll
  for (int f = 0; f < 4; ++f) {
    const int col = colbase + f * 16 + l16;
#pragma unroll
    for (int i = 0; i < 8; ++i)
      out[(size_t)(rbase + i) * DMODEL + col] = acc[f][i];
  }
}

// ---------------------------------------------------------------------------
// Host-side launcher
// ---------------------------------------------------------------------------
extern "C" void kernel_launch(void* const* d_in, const int* in_sizes, int n_in,
                              void* d_out, int out_size, void* d_ws, size_t ws_size,
                              hipStream_t stream) {
  const float* x    = (const float*)d_in[0];
  const float* wqkv = (const float*)d_in[1];
  const float* bqkv = (const float*)d_in[2];
  const float* wout = (const float*)d_in[3];
  const float* lsc  = (const float*)d_in[4];
  float* out = (float*)d_out;

  char* ws = (char*)d_ws;
  const size_t MB = 1024ull * 1024ull;
  bf16* xb    = (bf16*)(ws + 0);          // 16 MB
  bf16* wqb   = (bf16*)(ws + 16  * MB);   //  6 MB
  bf16* wob   = (bf16*)(ws + 22  * MB);   //  2 MB
  float* qf   = (float*)(ws + 24 * MB);   // 32 MB  (reused as ctx later)
  float* kf   = (float*)(ws + 56 * MB);   // 32 MB
  float* vf   = (float*)(ws + 88 * MB);   // 32 MB
  bf16* qb    = (bf16*)(ws + 120 * MB);   // 16 MB
  bf16* kb    = (bf16*)(ws + 136 * MB);   // 16 MB
  bf16* vT    = (bf16*)(ws + 152 * MB);   // 16 MB
  bf16* ctx   = (bf16*)(ws + 24 * MB);    // alias qf (dead after k_norm_pack)

  k_cvt_bf16<<<2048, 256, 0, stream>>>(x,    xb,  NROWS * DMODEL);
  k_cvt_bf16<<<2048, 256, 0, stream>>>(wqkv, wqb, QKV_N * DMODEL);
  k_cvt_bf16<<<1024, 256, 0, stream>>>(wout, wob, DMODEL * DMODEL);

  // QKV GEMM: 64 row-blocks x 48 col-groups
  k_qkv_gemm<<<3072, 256, 0, stream>>>(xb, wqb, bqkv, qf, kf, vf);

  // normalize + pack: B*H*N = 131072 rows / 8 waves per block
  k_norm_pack<<<16384, 256, 0, stream>>>(qf, kf, vf, lsc, qb, kb, vT);

  // attention: B*H*(N/16) = 8192 single-wave blocks
  k_attn<<<8192, 32, 0, stream>>>(qb, kb, vT, ctx);

  // output GEMM: 64 row-blocks x 16 col-groups
  k_out_gemm<<<1024, 256, 0, stream>>>(ctx, wob, out);
}
